// FlexCausalSelfAttention_55422257988307
// MI455X (gfx1250) — matrix operand order repaired and checked by reading, not compiled, not run on hardware
//
#include <hip/hip_runtime.h>

// ---------------------------------------------------------------------------
// MI455X (gfx1250) flex causal self-attention, bf16 WMMA pipeline.
// Compute-bound problem (~120 GFLOP vs ~150MB <-> 6.5us HBM @23.3TB/s), so
// everything matmul-shaped goes through v_wmma_f32_16x16x32_bf16, and the
// GEMM staging path uses CDNA5 async global->LDS copies (ASYNCcnt) with
// double-buffered LDS tiles so DMA overlaps the WMMA pipeline.
// ---------------------------------------------------------------------------

typedef __bf16 bf16_t;
typedef __attribute__((ext_vector_type(8)))  __bf16 v8bf;
typedef __attribute__((ext_vector_type(16))) __bf16 v16bf;
typedef __attribute__((ext_vector_type(8)))  float  v8f;
typedef __attribute__((ext_vector_type(4)))  int    v4i_t;

#define B_SZ  2
#define T_SZ  2048
#define C_SZ  2048
#define NH    16
#define NKV   4
#define HD    128
#define WIN   1024
#define SINKN 4
#define QSCALE 0.088388347648318447f   // 1/sqrt(128)

#define TM  128
#define TN  128
#define TKK 32

#ifndef __has_builtin
#define __has_builtin(x) 0
#endif

__device__ __forceinline__ v8f v8f_zero() {
  v8f z;
#pragma unroll
  for (int i = 0; i < 8; ++i) z[i] = 0.0f;
  return z;
}

__device__ __forceinline__ v16bf frag_cat(v8bf lo, v8bf hi) {
  return __builtin_shufflevector(lo, hi, 0,1,2,3,4,5,6,7,8,9,10,11,12,13,14,15);
}

// A-operand fragment (16x32 bf16, row-major source; caller passes this lane's
// row pointer). Per ISA 7.12.2: lanes 0-15 hold K 0..7 & 16..23, lanes 16-31
// hold K 8..15 & 24..31.
__device__ __forceinline__ v16bf load_frag_a(const bf16_t* rowp, int lane) {
  const int base8 = (lane >> 4) << 3;
  v8bf lo = *(const v8bf*)(rowp + base8);
  v8bf hi = *(const v8bf*)(rowp + base8 + 16);
  return frag_cat(lo, hi);
}

// B-operand fragment (32x16 bf16). Lane = output column; contraction dim is
// contiguous in memory at colp: lanes 0-15 hold K 0..7 & 16..23, lanes 16-31 K 8..15 & 24..31.
__device__ __forceinline__ v16bf load_frag_b(const bf16_t* colp, int lane) {
  const int base8 = (lane >> 4) << 3;
  v8bf lo = *(const v8bf*)(colp + base8);
  v8bf hi = *(const v8bf*)(colp + base8 + 16);
  return frag_cat(lo, hi);
}

__device__ __forceinline__ v8f wmma_bf16(v16bf a, v16bf b, v8f c) {
  return __builtin_amdgcn_wmma_f32_16x16x32_bf16(false, a, false, b,
                                                 (short)0, c, false, false);
}

// ---------------------------------------------------------------------------
// CDNA5 async global->LDS 16B copy (GLOBAL_LOAD_ASYNC_TO_LDS_B128, ASYNCcnt).
// Builtin expects (int4 addrspace(1)*, int4 addrspace(3)*, imm offset, cpol);
// generic-pointer low 32 bits are the wave-relative LDS byte address (ISA
// 10.2 aperture rules), so integer round-trips give us the right values.
// ---------------------------------------------------------------------------
__device__ __forceinline__ void async_copy16(void* lds_dst, const void* gsrc) {
#if __has_builtin(__builtin_amdgcn_global_load_async_to_lds_b128)
  __builtin_amdgcn_global_load_async_to_lds_b128(
      (__attribute__((address_space(1))) v4i_t*)(uintptr_t)gsrc,
      (__attribute__((address_space(3))) v4i_t*)(uint32_t)(uintptr_t)lds_dst,
      0, 0);
#else
  unsigned l = (unsigned)(uintptr_t)lds_dst;
  asm volatile("global_load_async_to_lds_b128 %0, %1, off"
               :: "v"(l), "v"(gsrc) : "memory");
#endif
}

__device__ __forceinline__ void async_wait0() {
#if __has_builtin(__builtin_amdgcn_s_wait_asynccnt)
  __builtin_amdgcn_s_wait_asynccnt(0);
#else
  asm volatile("s_wait_asynccnt 0" ::: "memory");
#endif
}

// ---------------------------------------------------------------------------
// fp32 -> bf16 conversion (vectorized 4-wide)
// ---------------------------------------------------------------------------
__global__ void cvt_f32_bf16(const float* __restrict__ s,
                             bf16_t* __restrict__ d, int n4) {
  int i = blockIdx.x * blockDim.x + threadIdx.x;
  if (i < n4) {
    float4 v = ((const float4*)s)[i];
    union { bf16_t b[4]; ushort4 u; } pk;
    pk.b[0] = (bf16_t)v.x; pk.b[1] = (bf16_t)v.y;
    pk.b[2] = (bf16_t)v.z; pk.b[3] = (bf16_t)v.w;
    ((ushort4*)d)[i] = pk.u;
  }
}

// ---------------------------------------------------------------------------
// Generic C(f32, MxN) = A(bf16, MxK row-major) x B(bf16, KxN row-major).
// 128x128 block tile, 8 waves (4x2), each wave 2x4 WMMA subtiles, K-step 32.
// Double-buffered LDS: A tile DMA'd via async-to-LDS, B tile loaded to
// registers early and transposed into LDS after the WMMA block so the
// s_wait_loadcnt lands behind 8 WMMAs. One barrier per K-step.
// ---------------------------------------------------------------------------
__global__ __launch_bounds__(256) void gemm_bf16f32(
    const bf16_t* __restrict__ A, const bf16_t* __restrict__ Bm,
    float* __restrict__ Cm, int M, int N, int K) {
  __shared__ __align__(16) bf16_t As[2][TM][TKK];   // 16 KB
  __shared__ __align__(16) bf16_t Bs[2][TN][TKK];   // 16 KB, Bs[n][k] = B[k][n]
  const int tid  = threadIdx.x;
  const int lane = tid & 31;
  const int w    = tid >> 5;   // 0..7
  const int wm   = w >> 1;     // 0..3
  const int wn   = w & 1;      // 0..1
  const int mrow = lane & 15;
  const int m0 = blockIdx.y * TM;
  const int n0 = blockIdx.x * TN;

  v8f acc[2][4];
#pragma unroll
  for (int i = 0; i < 2; ++i)
#pragma unroll
    for (int j = 0; j < 4; ++j) acc[i][j] = v8f_zero();

  // ---- prologue: stage tile 0 into buffer 0 ----
  {
    v8bf breg[2];
#pragma unroll
    for (int i = 0; i < 2; ++i) {              // B tile: 512 16B chunks / 256thr
      int c = tid + i * 256;
      int kk = c >> 4, nq = c & 15;
      breg[i] = *(const v8bf*)&Bm[(size_t)kk * N + n0 + nq * 8];
    }
#pragma unroll
    for (int i = 0; i < 2; ++i) {              // A tile: async DMA to LDS
      int c = tid + i * 256;
      int r = c >> 2, qd = c & 3;
      async_copy16(&As[0][r][qd * 8], &A[(size_t)(m0 + r) * K + qd * 8]);
    }
#pragma unroll
    for (int i = 0; i < 2; ++i) {              // transpose B into LDS
      int c = tid + i * 256;
      int kk = c >> 4, nq = c & 15;
#pragma unroll
      for (int e = 0; e < 8; ++e) Bs[0][nq * 8 + e][kk] = breg[i][e];
    }
    async_wait0();
    __syncthreads();
  }

  const int nk = K / TKK;
  for (int kt = 0; kt < nk; ++kt) {
    const int buf = kt & 1, nbuf = buf ^ 1;
    const int k0n = (kt + 1) * TKK;
    const bool have_next = (kt + 1) < nk;

    v8bf breg[2];
    if (have_next) {
      // issue next B tile global loads (consumed after the WMMA block)
#pragma unroll
      for (int i = 0; i < 2; ++i) {
        int c = tid + i * 256;
        int kk = c >> 4, nq = c & 15;
        breg[i] = *(const v8bf*)&Bm[(size_t)(k0n + kk) * N + n0 + nq * 8];
      }
      // next A tile: async DMA straight into the back LDS buffer
#pragma unroll
      for (int i = 0; i < 2; ++i) {
        int c = tid + i * 256;
        int r = c >> 2, qd = c & 3;
        async_copy16(&As[nbuf][r][qd * 8],
                     &A[(size_t)(m0 + r) * K + k0n + qd * 8]);
      }
      if (k0n + TKK < K)  // hint B tile k+2 toward the cache hierarchy
        __builtin_prefetch(&Bm[(size_t)(k0n + TKK) * N + n0 + (tid & 15) * 8],
                           0, 1);
    }

    // ---- compute on front buffer ----
    v16bf af[2], bfr[4];
#pragma unroll
    for (int sm = 0; sm < 2; ++sm)
      af[sm] = load_frag_a(&As[buf][wm * 32 + sm * 16 + mrow][0], lane);
#pragma unroll
    for (int sn = 0; sn < 4; ++sn)
      bfr[sn] = load_frag_b(&Bs[buf][wn * 64 + sn * 16 + mrow][0], lane);
#pragma unroll
    for (int sm = 0; sm < 2; ++sm)
#pragma unroll
      for (int sn = 0; sn < 4; ++sn)
        acc[sm][sn] = wmma_bf16(af[sm], bfr[sn], acc[sm][sn]);

    // ---- retire next B tile into the back buffer ----
    if (have_next) {
#pragma unroll
      for (int i = 0; i < 2; ++i) {
        int c = tid + i * 256;
        int kk = c >> 4, nq = c & 15;
#pragma unroll
        for (int e = 0; e < 8; ++e) Bs[nbuf][nq * 8 + e][kk] = breg[i][e];
      }
    }
    async_wait0();
    __syncthreads();
  }

  const int rhi  = (lane >> 4) << 3;
  const int ncol = lane & 15;
#pragma unroll
  for (int sm = 0; sm < 2; ++sm)
#pragma unroll
    for (int sn = 0; sn < 4; ++sn)
#pragma unroll
      for (int r = 0; r < 8; ++r) {
        int row = m0 + wm * 32 + sm * 16 + rhi + r;
        int col = n0 + wn * 64 + sn * 16 + ncol;
        Cm[(size_t)row * N + col] = acc[sm][sn][r];
      }
}

// ---------------------------------------------------------------------------
// RoPE + repack: Q -> [B,H,T,D] bf16 (scaled by 1/sqrt(D)),
// K -> [B,Hkv,T,D] bf16, V -> [B,Hkv,D,T] bf16 (transposed so the PV
// B-operand fragment is a contiguous load in the attention inner loop).
// ---------------------------------------------------------------------------
__global__ void rope_pack(const float* __restrict__ qf,
                          const float* __restrict__ kf,
                          const float* __restrict__ vf,
                          bf16_t* __restrict__ Qb, bf16_t* __restrict__ Kb,
                          bf16_t* __restrict__ Vt) {
  int idx = blockIdx.x * blockDim.x + threadIdx.x;
  const int total = B_SZ * T_SZ * NH * (HD / 2);
  if (idx >= total) return;
  int i = idx & 63;
  int h = (idx >> 6) & 15;
  int t = (idx >> 10) & 2047;
  int b = idx >> 21;

  float inv_freq = __powf(10000.0f, -(2.0f * (float)i) / 128.0f);
  float ang = (float)t * inv_freq;
  float cs = __cosf(ang), sn = __sinf(ang);

  const float* qp = qf + (size_t)(b * T_SZ + t) * C_SZ + h * HD;
  float x1 = qp[i], x2 = qp[i + 64];
  bf16_t* qo = Qb + ((size_t)(b * NH + h) * T_SZ + t) * HD;
  qo[i]      = (bf16_t)((x1 * cs - x2 * sn) * QSCALE);
  qo[i + 64] = (bf16_t)((x2 * cs + x1 * sn) * QSCALE);

  if (h < NKV) {
    const float* kp = kf + (size_t)(b * T_SZ + t) * (NKV * HD) + h * HD;
    float k1 = kp[i], k2 = kp[i + 64];
    bf16_t* ko = Kb + ((size_t)(b * NKV + h) * T_SZ + t) * HD;
    ko[i]      = (bf16_t)(k1 * cs - k2 * sn);
    ko[i + 64] = (bf16_t)(k2 * cs + k1 * sn);

    const float* vp = vf + (size_t)(b * T_SZ + t) * (NKV * HD) + h * HD;
    bf16_t* vo = Vt + (size_t)(b * NKV + h) * HD * T_SZ;
    vo[(size_t)i * T_SZ + t]        = (bf16_t)vp[i];
    vo[(size_t)(i + 64) * T_SZ + t] = (bf16_t)vp[i + 64];
  }
}

// ---------------------------------------------------------------------------
// Flash attention: one wave32 per 16-row Q tile. Scores via 4 WMMA (K=128),
// online softmax with 16-lane shfl_xor row reductions (wave32), P bounced
// through 1KB LDS (C-layout -> A-layout transpose), PV via 8 WMMA.
// ---------------------------------------------------------------------------
__global__ __launch_bounds__(32) void attn_fa(
    const bf16_t* __restrict__ Qb, const bf16_t* __restrict__ Kb,
    const bf16_t* __restrict__ Vt, const int* __restrict__ seq_lengths,
    bf16_t* __restrict__ Yb) {
  __shared__ __align__(16) bf16_t Pl[16][32];
  const int nqt = T_SZ / 16;
  int blk = blockIdx.x;
  int qt = blk % nqt;
  int h  = (blk / nqt) % NH;
  int b  = blk / (nqt * NH);
  int hv = h >> 2;   // H/Hkv = 4
  int lane = threadIdx.x & 31;
  int mrow = lane & 15;
  int rhi  = (lane >> 4) << 3;
  int L = seq_lengths[b]; if (L > T_SZ) L = T_SZ;
  int q0 = qt * 16;
  bf16_t* ybase = Yb + (size_t)b * T_SZ * C_SZ + h * HD;

  if (q0 >= L) {  // whole tile past valid length -> zero output rows
    int row = q0 + (lane >> 1);
    int c0  = (lane & 1) * 64;
    v8bf z;
#pragma unroll
    for (int e = 0; e < 8; ++e) z[e] = (bf16_t)0.0f;
#pragma unroll
    for (int e = 0; e < 8; ++e)
      *(v8bf*)(ybase + (size_t)row * C_SZ + c0 + e * 8) = z;
    return;
  }

  const bf16_t* qrow = Qb + ((size_t)(b * NH + h) * T_SZ + q0 + mrow) * HD;
  v16bf qfr[4];
#pragma unroll
  for (int c = 0; c < 4; ++c) qfr[c] = load_frag_a(qrow + 32 * c, lane);

  const bf16_t* kbp = Kb + (size_t)(b * NKV + hv) * T_SZ * HD;
  const bf16_t* vbp = Vt + (size_t)(b * NKV + hv) * HD * T_SZ;

  v8f o[8];
  float mrun[8], lrun[8];
#pragma unroll
  for (int j = 0; j < 8; ++j) o[j] = v8f_zero();
#pragma unroll
  for (int r = 0; r < 8; ++r) { mrun[r] = -1e30f; lrun[r] = 0.0f; }

  auto tile = [&](int kb) {
    v8f s[2];
    s[0] = v8f_zero(); s[1] = v8f_zero();
#pragma unroll
    for (int sb = 0; sb < 2; ++sb) {
      const bf16_t* krow = kbp + (size_t)(kb + sb * 16 + mrow) * HD;
#pragma unroll
      for (int c = 0; c < 4; ++c) {
        v16bf kfrag = load_frag_b(krow + 32 * c, lane);
        s[sb] = wmma_bf16(qfr[c], kfrag, s[sb]);
      }
    }
    float al[8];
#pragma unroll
    for (int r = 0; r < 8; ++r) {
      int qg  = q0 + rhi + r;
      int kg0 = kb + mrow;
      int kg1 = kg0 + 16;
      bool ok0 = (qg >= kg0) && ((qg - kg0 <= WIN) || (kg0 < SINKN)) &&
                 (kg0 < L) && (qg < L);
      bool ok1 = (qg >= kg1) && ((qg - kg1 <= WIN) || (kg1 < SINKN)) &&
                 (kg1 < L) && (qg < L);
      float s0 = ok0 ? s[0][r] : -1e30f;
      float s1 = ok1 ? s[1][r] : -1e30f;
      float t = fmaxf(s0, s1);
#pragma unroll
      for (int off = 1; off < 16; off <<= 1)
        t = fmaxf(t, __shfl_xor(t, off, 32));   // stays within 16-lane half
      float mnew = fmaxf(mrun[r], t);
      float p0 = __expf(s0 - mnew);
      float p1 = __expf(s1 - mnew);
      al[r] = __expf(mrun[r] - mnew);
      mrun[r] = mnew;
      Pl[rhi + r][mrow]      = (bf16_t)p0;   // C-layout -> LDS
      Pl[rhi + r][mrow + 16] = (bf16_t)p1;
      float rs = p0 + p1;
#pragma unroll
      for (int off = 1; off < 16; off <<= 1)
        rs += __shfl_xor(rs, off, 32);
      lrun[r] = lrun[r] * al[r] + rs;
    }
#pragma unroll
    for (int j = 0; j < 8; ++j)
#pragma unroll
      for (int r = 0; r < 8; ++r) o[j][r] *= al[r];
    __syncthreads();
    v16bf pf = load_frag_a(&Pl[mrow][0], lane);  // A-layout read-back
#pragma unroll
    for (int j = 0; j < 8; ++j) {
      const bf16_t* vrow = vbp + (size_t)(j * 16 + mrow) * T_SZ + kb;
      v16bf vfrag = load_frag_b(vrow, lane);
      o[j] = wmma_bf16(pf, vfrag, o[j]);
    }
    __syncthreads();
  };

  int wlo = (q0 > WIN) ? ((q0 - WIN) >> 5) : 0;
  int whi = (q0 + 15) >> 5;
  if (wlo > 0) tile(0);                        // sink tile (keys 0..31)
  for (int kt = wlo; kt <= whi; ++kt) tile(kt << 5);

#pragma unroll
  for (int r = 0; r < 8; ++r) {
    int qg = q0 + rhi + r;
    float inv = (qg < L && lrun[r] > 0.0f) ? (1.0f / lrun[r]) : 0.0f;
#pragma unroll
    for (int j = 0; j < 8; ++j)
      ybase[(size_t)qg * C_SZ + j * 16 + mrow] = (bf16_t)(o[j][r] * inv);
  }
}

// ---------------------------------------------------------------------------
// Host-side launch
// ---------------------------------------------------------------------------
extern "C" void kernel_launch(void* const* d_in, const int* in_sizes, int n_in,
                              void* d_out, int out_size, void* d_ws,
                              size_t ws_size, hipStream_t stream) {
  (void)in_sizes; (void)n_in; (void)out_size; (void)ws_size;
  const float* x  = (const float*)d_in[0];
  const float* Wq = (const float*)d_in[1];
  const float* Wk = (const float*)d_in[2];
  const float* Wv = (const float*)d_in[3];
  const float* Wo = (const float*)d_in[4];
  const int* seql = (const int*)d_in[5];
  float* out = (float*)d_out;

  const int M   = B_SZ * T_SZ;   // 4096
  const int KVC = NKV * HD;      // 512

  char* ws = (char*)d_ws;
  size_t off = 0;
  auto carve = [&](size_t bytes) -> void* {
    void* p = ws + off;
    off = (off + bytes + 255) & ~(size_t)255;
    return p;
  };
  bf16_t* xb  = (bf16_t*)carve((size_t)M * C_SZ * 2);
  bf16_t* wqb = (bf16_t*)carve((size_t)C_SZ * C_SZ * 2);
  bf16_t* wkb = (bf16_t*)carve((size_t)C_SZ * KVC * 2);
  bf16_t* wvb = (bf16_t*)carve((size_t)C_SZ * KVC * 2);
  bf16_t* wob = (bf16_t*)carve((size_t)C_SZ * C_SZ * 2);
  float*  qf  = (float*)carve((size_t)M * C_SZ * 4);
  float*  kf  = (float*)carve((size_t)M * KVC * 4);
  float*  vf  = (float*)carve((size_t)M * KVC * 4);
  bf16_t* Qb  = (bf16_t*)carve((size_t)M * C_SZ * 2);
  bf16_t* Kb  = (bf16_t*)carve((size_t)B_SZ * NKV * T_SZ * HD * 2);
  bf16_t* Vt  = (bf16_t*)carve((size_t)B_SZ * NKV * T_SZ * HD * 2);
  bf16_t* Yb  = (bf16_t*)carve((size_t)M * C_SZ * 2);

  auto cvt = [&](const float* s, bf16_t* d, size_t n) {
    int n4 = (int)(n / 4);
    cvt_f32_bf16<<<(n4 + 255) / 256, 256, 0, stream>>>(s, d, n4);
  };
  cvt(x,  xb,  (size_t)M * C_SZ);
  cvt(Wq, wqb, (size_t)C_SZ * C_SZ);
  cvt(Wk, wkb, (size_t)C_SZ * KVC);
  cvt(Wv, wvb, (size_t)C_SZ * KVC);
  cvt(Wo, wob, (size_t)C_SZ * C_SZ);

  dim3 blk(256);
  gemm_bf16f32<<<dim3(C_SZ / TN, M / TM), blk, 0, stream>>>(xb, wqb, qf, M, C_SZ, C_SZ);
  gemm_bf16f32<<<dim3(KVC  / TN, M / TM), blk, 0, stream>>>(xb, wkb, kf, M, KVC, C_SZ);
  gemm_bf16f32<<<dim3(KVC  / TN, M / TM), blk, 0, stream>>>(xb, wvb, vf, M, KVC, C_SZ);

  int rthreads = B_SZ * T_SZ * NH * (HD / 2);
  rope_pack<<<(rthreads + 255) / 256, 256, 0, stream>>>(qf, kf, vf, Qb, Kb, Vt);

  attn_fa<<<B_SZ * NH * (T_SZ / 16), 32, 0, stream>>>(Qb, Kb, Vt, seql, Yb);

  gemm_bf16f32<<<dim3(C_SZ / TN, M / TM), blk, 0, stream>>>(Yb, wob, out, M, C_SZ, C_SZ);
}
